// FFTConv2d_40166534152219
// MI455X (gfx1250) — matrix operand order repaired and checked by reading, not compiled, not run on hardware
//
#include <hip/hip_runtime.h>

// ---------------------------------------------------------------------------
// FFT-conv reference == 3x3 same-conv of [32,112,112,64] with [3,3,64,128],
// placed at circular offset 71 inside a [32,127,127,128] canvas.
// bf16 implicit GEMM with CDNA5 WMMA (wave32) + async global->LDS B staging.
//   M = 32*127*127 output pixels, N = 128 filters, K = 3*3*64 = 576.
// K is staged one full tap at a time (chunk = 64), 16 WMMAs per barrier.
// ---------------------------------------------------------------------------

typedef __bf16 v16bf __attribute__((ext_vector_type(16)));
typedef __bf16 v2bf  __attribute__((ext_vector_type(2)));
typedef float  v8f   __attribute__((ext_vector_type(8)));
typedef float  v2f   __attribute__((ext_vector_type(2)));

typedef int v4i __attribute__((vector_size(16)));            // matches builtin proto
typedef __attribute__((address_space(1))) v4i gv4i;          // global v4i
typedef __attribute__((address_space(3))) v4i lv4i;          // LDS v4i

#if defined(__has_builtin)
#  if __has_builtin(__builtin_amdgcn_global_load_async_to_lds_b128)
#    define USE_ASYNC_B 1
#  endif
#endif
#ifndef USE_ASYNC_B
#  define USE_ASYNC_B 0
#endif

static constexpr int H = 112, W = 112, CIN = 64, F = 128, NIMG = 32;
static constexpr int OUTD   = 127;                 // new_size - 1
static constexpr int OUT2   = OUTD * OUTD;         // 16129
static constexpr int TOTAL_PIX = NIMG * OUT2;      // 516128
static constexpr int KTAP   = 9;                   // 3x3 taps == K chunks
static constexpr int SHIFT  = 71;                  // fp(8) + kernel center(63)
static constexpr int ROW_U4 = 9;                   // 64 bf16 + 16B pad = 144B = 9 uint4

union V16 { uint4 u[2]; v16bf v; };

// f32x2 -> packed bf16x2 (targets v_cvt_pk_bf16_f32 on gfx1250)
__device__ static inline unsigned pk2_bf16(float a, float b) {
  v2f f = {a, b};
  return __builtin_bit_cast(unsigned, __builtin_convertvector(f, v2bf));
}

__device__ static inline gv4i* to_gv4(const void* p) {
  return (gv4i*)(uintptr_t)p;
}
__device__ static inline lv4i* to_lv4(void* p) {
  return (lv4i*)(unsigned)(uintptr_t)p;  // low 32 bits of generic = LDS offset
}

__device__ static inline void wait_async0() {
#if USE_ASYNC_B
#  if __has_builtin(__builtin_amdgcn_s_wait_asynccnt)
  __builtin_amdgcn_s_wait_asynccnt(0);
#  else
  asm volatile("s_wait_asynccnt 0" ::: "memory");
#  endif
#endif
}

struct AFrag { uint4 a[4]; };

// A staging: 32 f32 (contiguous 128B) of one zero-padded patch row -> bf16 regs.
// Chunk tap == (kh,kw); this thread covers channels [hlf*32, hlf*32+32).
__device__ static inline AFrag load_a(int tap, const float* __restrict__ in,
                                      bool pixValid, int n, int i, int j,
                                      int hlf) {
  AFrag f;
  const int kh   = tap / 3, kw = tap % 3;
  const int cin0 = hlf * 32;
  float4 v[8];
  for (int r = 0; r < 8; ++r) v[r] = make_float4(0.f, 0.f, 0.f, 0.f);
  const int ih = i + kh - 1, iw = j + kw - 1;
  if (pixValid && (unsigned)ih < (unsigned)H && (unsigned)iw < (unsigned)W) {
    const float4* src = reinterpret_cast<const float4*>(
        in + (((n * H + ih) * W + iw) * CIN + cin0));
    for (int r = 0; r < 8; ++r) v[r] = src[r];
  }
  for (int r = 0; r < 4; ++r)
    f.a[r] = make_uint4(pk2_bf16(v[2*r].x,   v[2*r].y),
                        pk2_bf16(v[2*r].z,   v[2*r].w),
                        pk2_bf16(v[2*r+1].x, v[2*r+1].y),
                        pk2_bf16(v[2*r+1].z, v[2*r+1].w));
  return f;
}

// B staging: 64 bytes of pre-transposed bf16 weights -> LDS (async, ASYNCcnt).
#if USE_ASYNC_B
__device__ static inline void stage_b_async(int tap, uint4* sBbuf,
                                            const unsigned short* __restrict__ wt,
                                            int p, int hlf) {
  const unsigned short* g = wt + tap * (F * CIN) + p * CIN + hlf * 32;
  uint4* l = sBbuf + p * ROW_U4 + hlf * 4;
  // IOFFSET applies to BOTH global and LDS addresses (ISA async LDS ops).
  __builtin_amdgcn_global_load_async_to_lds_b128(to_gv4(g), to_lv4(l), 0,  0);
  __builtin_amdgcn_global_load_async_to_lds_b128(to_gv4(g), to_lv4(l), 16, 0);
  __builtin_amdgcn_global_load_async_to_lds_b128(to_gv4(g), to_lv4(l), 32, 0);
  __builtin_amdgcn_global_load_async_to_lds_b128(to_gv4(g), to_lv4(l), 48, 0);
}
#else
struct BFrag { uint4 b[4]; };
__device__ static inline BFrag load_b(int tap, const unsigned short* __restrict__ wt,
                                      int p, int hlf) {
  const uint4* bsrc = reinterpret_cast<const uint4*>(
      wt + tap * (F * CIN) + p * CIN + hlf * 32);
  BFrag f;
  for (int r = 0; r < 4; ++r) f.b[r] = bsrc[r];
  return f;
}
#endif

// Weights [3][3][64][128] f32 -> bf16 [tap][F][CIN] so B staging is contiguous.
__global__ void __launch_bounds__(256)
wprep_kernel(const float* __restrict__ w, unsigned short* __restrict__ wt) {
  int idx = blockIdx.x * 256 + threadIdx.x;
  if (idx >= KTAP * CIN * F) return;
  int tap = idx / (CIN * F);
  int r   = idx - tap * (CIN * F);
  int cin = r / F;
  int f   = r - cin * F;
  v2f x = {w[idx], 0.0f};
  unsigned packed = __builtin_bit_cast(unsigned, __builtin_convertvector(x, v2bf));
  wt[tap * (F * CIN) + f * CIN + cin] = (unsigned short)(packed & 0xFFFFu);
}

__global__ void __launch_bounds__(256)
fftconv_wmma_kernel(const float* __restrict__ in,
                    const unsigned short* __restrict__ wt,
                    float* __restrict__ out) {
  // Double-buffered bf16 tiles: A[128 pix][64 k], B[128 filt][64 k], padded rows.
  // 4 tiles x 18KB = 72KB of the 320KB WGP LDS.
  __shared__ uint4 sA[2][128 * ROW_U4];
  __shared__ uint4 sB[2][128 * ROW_U4];

  const int t    = threadIdx.x;
  const int lane = t & 31, wv = t >> 5;
  const int wm   = wv & 3, wn = wv >> 2;        // wave grid: 4 (M) x 2 (N)
  const int p    = t >> 1, hlf = t & 1;         // staging: row p, 32-elem half
  const long tileBase = (long)blockIdx.x * 128;

  // Map this thread's staging row to an input pixel (canvas -> conv coords).
  int n = 0, i = 0, j = 0; bool pixValid = false;
  {
    long P = tileBase + p;
    if (P < TOTAL_PIX) {
      n = (int)(P / OUT2);
      int rem = (int)(P - (long)n * OUT2);
      int oh = rem / OUTD, ow = rem - oh * OUTD;
      i = oh - SHIFT; if (i < 0) i += OUTD;
      j = ow - SHIFT; if (j < 0) j += OUTD;
      pixValid = (i < H) && (j < W);            // outside -> zeros in canvas
    }
  }

  const v8f vzero = {0,0,0,0,0,0,0,0};
  v8f acc[2][4];
  for (int mi = 0; mi < 2; ++mi)
    for (int nj = 0; nj < 4; ++nj) acc[mi][nj] = vzero;

  // Prologue: stage tap 0.
  AFrag cur = load_a(0, in, pixValid, n, i, j, hlf);
  for (int r = 0; r < 4; ++r) sA[0][p * ROW_U4 + hlf * 4 + r] = cur.a[r];
#if USE_ASYNC_B
  stage_b_async(0, sB[0], wt, p, hlf);
#else
  BFrag bcur = load_b(0, wt, p, hlf);
  for (int r = 0; r < 4; ++r) sB[0][p * ROW_U4 + hlf * 4 + r] = bcur.b[r];
#endif

  const int lm = lane & 15, kh2 = lane >> 4;    // fragment lane decomposition

  for (int kc = 0; kc < KTAP; ++kc) {
    wait_async0();                // our async-to-LDS writes landed
    __syncthreads();              // everyone's tile (DS + async) is visible
    const bool more = (kc + 1 < KTAP);
    if (more) {
      cur = load_a(kc + 1, in, pixValid, n, i, j, hlf);
#if USE_ASYNC_B
      stage_b_async(kc + 1, sB[(kc + 1) & 1], wt, p, hlf);  // overlaps WMMA below
#else
      bcur = load_b(kc + 1, wt, p, hlf);
#endif
    }

    const uint4* A = sA[kc & 1];
    const uint4* B = sB[kc & 1];

    // Two K=32 WMMA steps over the staged K=64 tap.
    for (int s = 0; s < 2; ++s) {
      // A frag (16x32): lane = row M=lm; K groups [kh2*8..+7], [16+kh2*8..+7]
      v16bf afr[2];
      for (int mi = 0; mi < 2; ++mi) {
        const int m = wm * 32 + mi * 16 + lm;
        V16 u;
        u.u[0] = A[m * ROW_U4 + s * 4 + kh2];
        u.u[1] = A[m * ROW_U4 + s * 4 + 2 + kh2];
        afr[mi] = u.v;
      }
      // B frag (32x16): lane = col N=lm; K groups [kh2*8..+7], [16+kh2*8..+7]
      v16bf bfr[4];
      for (int nj = 0; nj < 4; ++nj) {
        const int nn = wn * 64 + nj * 16 + lm;
        V16 u;
        u.u[0] = B[nn * ROW_U4 + s * 4 + kh2];
        u.u[1] = B[nn * ROW_U4 + s * 4 + 2 + kh2];
        bfr[nj] = u.v;
      }
      for (int mi = 0; mi < 2; ++mi)
        for (int nj = 0; nj < 4; ++nj)
          acc[mi][nj] = __builtin_amdgcn_wmma_f32_16x16x32_bf16(
              false, afr[mi], false, bfr[nj], (short)0, acc[mi][nj], false, false);
    }

    if (more) {
      const int b = (kc + 1) & 1;
      for (int r = 0; r < 4; ++r) sA[b][p * ROW_U4 + hlf * 4 + r] = cur.a[r];
#if !USE_ASYNC_B
      for (int r = 0; r < 4; ++r) sB[b][p * ROW_U4 + hlf * 4 + r] = bcur.b[r];
#endif
    }
  }

  // Writeout: C/D layout -> lanes 0-15: N=lane,M=vgpr; lanes 16-31: N=lane-16,M=vgpr+8
  for (int mi = 0; mi < 2; ++mi) {
    for (int v = 0; v < 8; ++v) {
      const long Pr = tileBase + wm * 32 + mi * 16 + kh2 * 8 + v;
      if (Pr < TOTAL_PIX) {
        float* o = out + Pr * F + wn * 64 + lm;
        o[0]  = acc[mi][0][v];
        o[16] = acc[mi][1][v];
        o[32] = acc[mi][2][v];
        o[48] = acc[mi][3][v];
      }
    }
  }
}

extern "C" void kernel_launch(void* const* d_in, const int* in_sizes, int n_in,
                              void* d_out, int out_size, void* d_ws, size_t ws_size,
                              hipStream_t stream) {
  const float* in = (const float*)d_in[0];     // [32,112,112,64] f32
  const float* w  = (const float*)d_in[1];     // [3,3,64,128]    f32
  float* out = (float*)d_out;                  // [32,127,127,128] f32
  unsigned short* wt = (unsigned short*)d_ws;  // bf16 [9][128][64] = 147 KB

  wprep_kernel<<<(KTAP * CIN * F + 255) / 256, 256, 0, stream>>>(w, wt);

  const int grid = (TOTAL_PIX + 127) / 128;    // 4033 M-tiles of 128 pixels
  fftconv_wmma_kernel<<<grid, 256, 0, stream>>>(in, wt, out);
}
